// RSNN_32220844654715
// MI455X (gfx1250) — matrix operand order repaired and checked by reading, not compiled, not run on hardware
//
#include <hip/hip_runtime.h>

// ---------------------------------------------------------------------------
// RSNN forward for MI455X (gfx1250, wave32, WMMA).
//
// Precision: split-bf16 ("3-product") emulation of fp32 GEMMs on
// v_wmma_f32_16x16x32_bf16 (f32 accumulation): x@W ~= xhi@Whi + xhi@Wlo +
// xlo@Whi, residual ~2^-16 relative. Spikes are exactly representable in
// bf16, so the recurrent GEMM needs only spk@Whi + spk@Wlo (weight exact).
//
// Roofline: ~460 MB of unavoidable HBM traffic (~20us @ 23.3 TB/s); bf16
// WMMA keeps the 55 GFLOP (3x18.4) input GEMM near that bound, where fp32
// WMMA (16x16x4) would be ~8x slower per instruction.
// ---------------------------------------------------------------------------

typedef __attribute__((ext_vector_type(16))) __bf16 v16bf;
typedef __attribute__((ext_vector_type(8)))  float  v8f;
typedef unsigned short u16;
typedef unsigned int   u32;

union BF16x16 {      // one wave-lane's A or B fragment (16 bf16 = 32 bytes)
  v16bf v;
  u16   u[16];
  uint4 q[2];
};

__device__ __forceinline__ u16 f2bf(float f) {      // RNE float -> bf16
  u32 x = __float_as_uint(f);
  x += 0x7FFFu + ((x >> 16) & 1u);
  return (u16)(x >> 16);
}
__device__ __forceinline__ float bf2f(u16 h) {
  return __uint_as_float(((u32)h) << 16);
}

__device__ __forceinline__ v8f wmma_bf16(v16bf a, v16bf b, v8f c) {
  // D(f32 16x16) = A(bf16 16x32) * B(bf16 32x16) + C
  return __builtin_amdgcn_wmma_f32_16x16x32_bf16(
      /*neg_a=*/false, a, /*neg_b=*/false, b,
      /*c_mod=*/(short)0, c, /*reuse_a=*/false, /*reuse_b=*/false);
}

constexpr int Bb = 512, Tt = 100, Ii = 700, Hh = 256;
constexpr int WIN_STRIDE = 704;   // W_in rows padded to 704 (16B aligned, zero tail)
constexpr float BETA = 0.9f;

// ---------------------------------------------------------------------------
// Kernel 1: split W_in / W_rec into bf16 hi/lo planes (tiny, one shot).
// ---------------------------------------------------------------------------
__global__ void prep_weights(const float* __restrict__ Win,
                             const float* __restrict__ Wrec,
                             u16* __restrict__ WinHi, u16* __restrict__ WinLo,
                             u16* __restrict__ WrecHi, u16* __restrict__ WrecLo) {
  const int NIN  = Hh * WIN_STRIDE;   // 180224 (padded)
  const int NREC = Hh * Hh;           // 65536
  int i = blockIdx.x * blockDim.x + threadIdx.x;
  if (i < NIN) {
    int n = i / WIN_STRIDE, k = i - n * WIN_STRIDE;
    float f = (k < Ii) ? Win[n * Ii + k] : 0.0f;   // zero-filled K tail
    u16 hi = f2bf(f);
    WinHi[i] = hi;
    WinLo[i] = f2bf(f - bf2f(hi));
  } else if (i < NIN + NREC) {
    int j = i - NIN;
    float f = Wrec[j];                 // diagonal already zeroed by harness
    u16 hi = f2bf(f);
    WrecHi[j] = hi;
    WrecLo[j] = f2bf(f - bf2f(hi));
  }
}

// ---------------------------------------------------------------------------
// Kernel 2: I_all[M=51200, H=256] = x[M, 700] @ W_in^T + b_in  (split-bf16).
// Block = 256 threads = 8 waves. Wave owns a 16(M) x 128(N) tile:
// A fragment converted once per K-slice, reused across 8 N-tiles x 3 WMMAs.
// ---------------------------------------------------------------------------
__global__ __launch_bounds__(256) void input_gemm(
    const float* __restrict__ x, const u16* __restrict__ WinHi,
    const u16* __restrict__ WinLo, const float* __restrict__ b_in,
    float* __restrict__ I_all) {
  const int lane = threadIdx.x & 31;
  const int wave = threadIdx.x >> 5;
  const int half = lane >> 4;        // A/B K-half per ISA fragment layout
  const int l16  = lane & 15;

  const int m_base = blockIdx.x * 64 + (wave >> 1) * 16;  // grid.x = 800
  const int n_base = (wave & 1) * 128;

  v8f acc[8];
  #pragma unroll
  for (int t = 0; t < 8; ++t)
    #pragma unroll
    for (int r = 0; r < 8; ++r) acc[t][r] = 0.0f;

  const float* xrow = x + (size_t)(m_base + l16) * Ii;   // A: row = l16 both halves

  for (int k0 = 0; k0 < Ii; k0 += 32) {                  // 22 K-slices
    // ---- A fragment: 16 f32 per lane -> bf16 hi/lo ----
    // layout: lanes 0-15 hold K = k0+{0..7, 16..23}; lanes 16-31 +8
    float af[16];
    const int kc = k0 + half * 8;
    if (k0 + 32 <= Ii) {             // fast path, fully in-bounds, 16B aligned
      float4 q0 = *(const float4*)(xrow + kc);
      float4 q1 = *(const float4*)(xrow + kc + 4);
      float4 q2 = *(const float4*)(xrow + kc + 16);
      float4 q3 = *(const float4*)(xrow + kc + 20);
      af[0]=q0.x; af[1]=q0.y; af[2]=q0.z;  af[3]=q0.w;
      af[4]=q1.x; af[5]=q1.y; af[6]=q1.z;  af[7]=q1.w;
      af[8]=q2.x; af[9]=q2.y; af[10]=q2.z; af[11]=q2.w;
      af[12]=q3.x; af[13]=q3.y; af[14]=q3.z; af[15]=q3.w;
    } else {                         // K tail (k0 = 672)
      #pragma unroll
      for (int e = 0; e < 16; ++e) {
        int k = kc + (e & 7) + ((e >> 3) << 4);
        af[e] = (k < Ii) ? xrow[k] : 0.0f;
      }
    }
    BF16x16 ahi, alo;
    #pragma unroll
    for (int e = 0; e < 16; ++e) {
      u16 h = f2bf(af[e]);
      ahi.u[e] = h;
      alo.u[e] = f2bf(af[e] - bf2f(h));
    }

    // ---- B fragments per N-tile, 3-product accumulate ----
    // layout: lane n = l16; lanes 0-15 K = kb+{0..7, 16..23} (kb = k0), lanes 16-31 +8
    const int kb = k0 + half * 8;    // padded rows: kb+23 always in-bounds
    #pragma unroll
    for (int nt = 0; nt < 8; ++nt) {
      const int n = n_base + nt * 16 + l16;
      const u16* ph = WinHi + (size_t)n * WIN_STRIDE + kb;
      const u16* pl = WinLo + (size_t)n * WIN_STRIDE + kb;
      BF16x16 bhi, blo;
      bhi.q[0] = *(const uint4*)(ph);
      bhi.q[1] = *(const uint4*)(ph + 16);
      blo.q[0] = *(const uint4*)(pl);
      blo.q[1] = *(const uint4*)(pl + 16);
      acc[nt] = wmma_bf16(ahi.v, bhi.v, acc[nt]);
      acc[nt] = wmma_bf16(ahi.v, blo.v, acc[nt]);
      acc[nt] = wmma_bf16(alo.v, bhi.v, acc[nt]);
    }
  }

  // ---- store D + b_in;  D layout: M = r + 8*half, N = l16 ----
  #pragma unroll
  for (int nt = 0; nt < 8; ++nt) {
    const int n = n_base + nt * 16 + l16;
    const float bias = b_in[n];
    #pragma unroll
    for (int r = 0; r < 8; ++r) {
      const int m = m_base + r + half * 8;
      I_all[(size_t)m * Hh + n] = acc[nt][r] + bias;
    }
  }
}

// ---------------------------------------------------------------------------
// Kernel 3: LIF scan. Batch rows are independent -> 32 workgroups x 16 rows.
// 512 threads = 16 waves; wave w owns columns [16w, 16w+16) of its 16 rows.
// mem state stays in VGPRs for all 100 steps; spikes exchanged via LDS
// (bf16, row-padded to 264 to avoid 16-way bank conflicts on the A gather).
// Recurrent product: spk(bf16 exact) @ (Whi + Wlo) via 2 WMMAs per K-slice.
// ---------------------------------------------------------------------------
__global__ __launch_bounds__(512) void lif_scan(
    const float* __restrict__ I_all, const u16* __restrict__ WrecHi,
    const u16* __restrict__ WrecLo, const float* __restrict__ b_rec,
    float* __restrict__ out) {
  constexpr int LP = 264;                     // LDS row pitch (ushorts)
  __shared__ u16 spkLDS[16][LP];              // spikes of this block's 16 rows

  const int lane = threadIdx.x & 31;
  const int wave = threadIdx.x >> 5;          // 0..15
  const int half = lane >> 4;
  const int l16  = lane & 15;

  const int b0 = blockIdx.x * 16;             // grid.x = 32
  const int n  = wave * 16 + l16;             // this lane's H column
  const float br = b_rec[n];

  const size_t MEMOFF = (size_t)Bb * Tt * Hh; // out = [spk | mem]

  v8f mem;
  #pragma unroll
  for (int r = 0; r < 8; ++r) mem[r] = 0.0f;

  // ---- t = 0: input current only (reference skips recurrent layer + b_rec)
  #pragma unroll
  for (int r = 0; r < 8; ++r) {
    const int b = b0 + r + half * 8;
    const size_t base = ((size_t)b * Tt + 0) * Hh;
    mem[r] = I_all[base + n];                 // beta*0 + cur - 0
    const float s = (mem[r] > 1.0f) ? 1.0f : 0.0f;
    out[base + n]          = s;
    out[MEMOFF + base + n] = mem[r];
    spkLDS[r + half * 8][n] = (s > 0.5f) ? (u16)0x3F80 : (u16)0;  // bf16 1/0
  }
  __syncthreads();

  for (int t = 1; t < Tt; ++t) {
    // Gather A fragments: spk(t-1) of our 16 rows, K = all 256 columns.
    BF16x16 A[8];
    #pragma unroll
    for (int s = 0; s < 8; ++s) {
      const int koff = s * 32 + half * 16;
      A[s].q[0] = *(const uint4*)&spkLDS[l16][koff];
      A[s].q[1] = *(const uint4*)&spkLDS[l16][koff + 8];
    }
    __syncthreads();                          // reads done before spk(t) writes

    v8f acc;
    #pragma unroll
    for (int r = 0; r < 8; ++r) acc[r] = 0.0f;

    #pragma unroll
    for (int s = 0; s < 8; ++s) {             // K = 256 in 8 slices of 32
      const int kb = s * 32 + half * 16;
      const u16* ph = WrecHi + (size_t)n * Hh + kb;   // L2-resident (128 KB)
      const u16* pl = WrecLo + (size_t)n * Hh + kb;
      BF16x16 bhi, blo;
      bhi.q[0] = *(const uint4*)(ph);
      bhi.q[1] = *(const uint4*)(ph + 8);
      blo.q[0] = *(const uint4*)(pl);
      blo.q[1] = *(const uint4*)(pl + 8);
      acc = wmma_bf16(A[s].v, bhi.v, acc);    // spk exact in bf16 ->
      acc = wmma_bf16(A[s].v, blo.v, acc);    // hi+lo reconstructs fp32 W
    }

    // LIF update (snntorch Leaky, reset-by-subtraction, THETA = 1)
    #pragma unroll
    for (int r = 0; r < 8; ++r) {
      const int b = b0 + r + half * 8;
      const size_t base = ((size_t)b * Tt + t) * Hh;
      const float cur = I_all[base + n] + acc[r] + br;
      const float rst = (mem[r] > 1.0f) ? 1.0f : 0.0f;   // from incoming mem
      mem[r] = BETA * mem[r] + cur - rst;
      const float s = (mem[r] > 1.0f) ? 1.0f : 0.0f;
      out[base + n]          = s;
      out[MEMOFF + base + n] = mem[r];
      spkLDS[r + half * 8][n] = (s > 0.5f) ? (u16)0x3F80 : (u16)0;
    }
    __syncthreads();                          // spk(t) visible for next step
  }
}

// ---------------------------------------------------------------------------
extern "C" void kernel_launch(void* const* d_in, const int* in_sizes, int n_in,
                              void* d_out, int out_size, void* d_ws, size_t ws_size,
                              hipStream_t stream) {
  (void)in_sizes; (void)n_in; (void)out_size;

  const float* x     = (const float*)d_in[0];   // [512,100,700]
  const float* W_in  = (const float*)d_in[1];   // [256,700]
  const float* b_in  = (const float*)d_in[2];   // [256]
  const float* W_rec = (const float*)d_in[3];   // [256,256], zero diagonal
  const float* b_rec = (const float*)d_in[4];   // [256]
  float* out = (float*)d_out;                   // [spk(512,100,256) | mem(...)]

  // Workspace layout (~53.4 MB total)
  char* ws = (char*)d_ws;
  size_t off = 0;
  float* I_all  = (float*)(ws + off); off += (size_t)Bb * Tt * Hh * 4;   // 52.4 MB
  u16* WinHi    = (u16*)(ws + off);   off += (size_t)Hh * WIN_STRIDE * 2;
  u16* WinLo    = (u16*)(ws + off);   off += (size_t)Hh * WIN_STRIDE * 2;
  u16* WrecHi   = (u16*)(ws + off);   off += (size_t)Hh * Hh * 2;
  u16* WrecLo   = (u16*)(ws + off);   off += (size_t)Hh * Hh * 2;
  (void)ws_size;

  const int prep_n = Hh * WIN_STRIDE + Hh * Hh;           // 245760
  prep_weights<<<dim3((prep_n + 255) / 256), dim3(256), 0, stream>>>(
      W_in, W_rec, WinHi, WinLo, WrecHi, WrecLo);

  input_gemm<<<dim3((Bb * Tt) / 64), dim3(256), 0, stream>>>(
      x, WinHi, WinLo, b_in, I_all);                       // 800 blocks

  lif_scan<<<dim3(Bb / 16), dim3(512), 0, stream>>>(
      I_all, WrecHi, WrecLo, b_rec, out);                  // 32 blocks
}
